// SimpleVectorQuantizer_37821482009268
// MI455X (gfx1250) — matrix operand order repaired and checked by reading, not compiled, not run on hardware
//
#include <hip/hip_runtime.h>

typedef __attribute__((ext_vector_type(16))) __bf16 v16bf;
typedef __attribute__((ext_vector_type(8)))  __bf16 v8bf;
typedef __attribute__((ext_vector_type(8)))  float  v8f;

#define NQ    65536        // B*R*C = 16*32*128
#define KDIM  64
#define SCB   512
#define QT    4            // query tiles (of 16) per wave
#define WPB   8            // waves per block
#define BLOCK (WPB * 32)
#define QPB   (WPB * QT * 16)           // 512 queries per block
#define NWAVES (NQ / (QT * 16))         // 1024 waves total

static const size_t OFF_VHAT = 0;
static const size_t OFF_Z    = (size_t)NQ * KDIM;   // 4194304
static const size_t OFF_LC   = OFF_Z + NQ;          // 4259840
static const size_t OFF_LB   = OFF_LC + 1;

union BV16 { v16bf v; v8bf h[2]; __bf16 e[16]; };

// ---------------------------------------------------------------------------
// Prep: per-codeword squared norms (fp32, exact)
// ---------------------------------------------------------------------------
__global__ void vq_prep(const float* __restrict__ cb, float* __restrict__ cnorm2) {
    int s = blockIdx.x * blockDim.x + threadIdx.x;
    if (s < SCB) {
        const float* row = cb + (size_t)s * KDIM;
        float t = 0.f;
        #pragma unroll 8
        for (int k = 0; k < KDIM; ++k) t = fmaf(row[k], row[k], t);
        cnorm2[s] = t;
    }
}

// ---------------------------------------------------------------------------
// Main: split-bf16 WMMA distance GEMM + fused argmin + gather + loss partials
// ---------------------------------------------------------------------------
__global__ void __launch_bounds__(BLOCK, 1)
vq_main(const float* __restrict__ vecs, const float* __restrict__ cb,
        const float* __restrict__ cnorm2, float* __restrict__ partial,
        float* __restrict__ vhat, float* __restrict__ zf)
{
    extern __shared__ char smem[];
    __bf16* lds_hi = (__bf16*)smem;                 // 512*64 bf16 = 64 KB
    __bf16* lds_lo = lds_hi + SCB * KDIM;           // 64 KB
    float*  lds_cn = (float*)(lds_lo + SCB * KDIM); // 2 KB

    // Stage codebook into LDS as hi/lo bf16 split (one pass per workgroup)
    for (int i = threadIdx.x; i < SCB * KDIM; i += BLOCK) {
        float c = cb[i];
        __bf16 hb = (__bf16)c;
        lds_hi[i] = hb;
        lds_lo[i] = (__bf16)(c - (float)hb);
    }
    for (int i = threadIdx.x; i < SCB; i += BLOCK) lds_cn[i] = cnorm2[i];
    __syncthreads();

    const int wave = threadIdx.x >> 5;
    const int lane = threadIdx.x & 31;
    const int ln   = lane & 15;
    const int half = lane >> 4;
    const int qbase0 = (blockIdx.x * WPB + wave) * (QT * 16);

    // Load + split QT query tiles (B operand: col n = lane%16, K contiguous per half)
    v16bf bh[QT][2], bl[QT][2];
    float vn2[QT];
    #pragma unroll
    for (int qt = 0; qt < QT; ++qt) {
        const float* vrow = vecs + (size_t)(qbase0 + qt * 16 + ln) * KDIM;
        float p = 0.f;
        #pragma unroll
        for (int c = 0; c < 2; ++c) {
            const float4* pv = (const float4*)(vrow + c * 32 + half * 16);
            float f[16];
            ((float4*)f)[0] = pv[0]; ((float4*)f)[1] = pv[1];
            ((float4*)f)[2] = pv[2]; ((float4*)f)[3] = pv[3];
            BV16 H, L;
            #pragma unroll
            for (int i = 0; i < 16; ++i) {
                __bf16 hb = (__bf16)f[i];
                H.e[i] = hb;
                L.e[i] = (__bf16)(f[i] - (float)hb);
                p = fmaf(f[i], f[i], p);
            }
            bh[qt][c] = H.v;
            bl[qt][c] = L.v;
        }
        vn2[qt] = p + __shfl_xor(p, 16);  // lane pair covers the full K=64
    }

    float bestKey[QT];
    int   bestIdx[QT];
    #pragma unroll
    for (int qt = 0; qt < QT; ++qt) { bestKey[qt] = 3.0e38f; bestIdx[qt] = 0; }

    // Sweep all 32 candidate tiles
    for (int t = 0; t < SCB / 16; ++t) {
        const int row = t * 16 + ln;
        BV16 AH[2], AL[2];
        #pragma unroll
        for (int c = 0; c < 2; ++c) {  // A pattern: k = c*32 + half*16 + {0..15}
            const __bf16* ph = lds_hi + row * KDIM + c * 32 + half * 16;
            const __bf16* pl = lds_lo + row * KDIM + c * 32 + half * 16;
            AH[c].h[0] = *(const v8bf*)ph;
            AH[c].h[1] = *(const v8bf*)(ph + 8);
            AL[c].h[0] = *(const v8bf*)pl;
            AL[c].h[1] = *(const v8bf*)(pl + 8);
        }
        float cn[8];
        #pragma unroll
        for (int j = 0; j < 8; ++j) cn[j] = lds_cn[t * 16 + 8 * half + j];

        #pragma unroll
        for (int qt = 0; qt < QT; ++qt) {
            v8f acc = {};
            #pragma unroll
            for (int c = 0; c < 2; ++c) {
                acc = __builtin_amdgcn_wmma_f32_16x16x32_bf16(false, AH[c].v, false, bh[qt][c], (short)0, acc, false, false);
                acc = __builtin_amdgcn_wmma_f32_16x16x32_bf16(false, AH[c].v, false, bl[qt][c], (short)0, acc, false, false);
                acc = __builtin_amdgcn_wmma_f32_16x16x32_bf16(false, AL[c].v, false, bh[qt][c], (short)0, acc, false, false);
            }
            #pragma unroll
            for (int j = 0; j < 8; ++j) {         // key = ||c||^2 - 2 v.c
                float key = fmaf(-2.0f, acc[j], cn[j]);
                int   m   = t * 16 + 8 * half + j;
                bool  b   = key < bestKey[qt];    // strict <: first-min within lane subset
                bestKey[qt] = b ? key : bestKey[qt];
                bestIdx[qt] = b ? m   : bestIdx[qt];
            }
        }
    }

    // Combine halves (tie -> lower index, matching jnp.argmin), write z/vhat/loss
    float wsum = 0.f;
    #pragma unroll
    for (int qt = 0; qt < QT; ++qt) {
        float ok = __shfl_xor(bestKey[qt], 16);
        int   oi = __shfl_xor(bestIdx[qt], 16);
        bool take = (ok < bestKey[qt]) || ((ok == bestKey[qt]) && (oi < bestIdx[qt]));
        float fk = take ? ok : bestKey[qt];
        int   zi = take ? oi : bestIdx[qt];
        float e2 = fmaxf(vn2[qt] + fk, 0.f);      // relu(||v||^2 - 2 v.c + ||c||^2)
        if (half == 0) zf[qbase0 + qt * 16 + ln] = (float)zi;
        wsum += e2;                               // duplicated in both halves

        for (int q = 0; q < 16; ++q) {            // gather: vhat[n,:] = codebook[z,:]
            int zq = __shfl(zi, q);
            const float2* src = (const float2*)(cb + (size_t)zq * KDIM);
            float2* dst = (float2*)(vhat + (size_t)(qbase0 + qt * 16 + q) * KDIM);
            dst[lane] = src[lane];
        }
    }
    #pragma unroll
    for (int off = 16; off > 0; off >>= 1) wsum += __shfl_xor(wsum, off);
    if (lane == 0) partial[blockIdx.x * WPB + wave] = 0.5f * wsum;  // /2: halves duplicate
}

// ---------------------------------------------------------------------------
// Finalize: deterministic reduction of per-wave partials -> both loss scalars
// ---------------------------------------------------------------------------
__global__ void vq_fin(const float* __restrict__ partial, float* __restrict__ out) {
    int lane = threadIdx.x;
    float s = 0.f;
    for (int i = lane; i < NWAVES; i += 32) s += partial[i];
    #pragma unroll
    for (int off = 16; off > 0; off >>= 1) s += __shfl_xor(s, off);
    if (lane == 0) {
        float m = s * (1.0f / (float)NQ);
        out[OFF_LC] = m;
        out[OFF_LB] = m;   // commit and codebook losses are numerically identical
    }
}

extern "C" void kernel_launch(void* const* d_in, const int* in_sizes, int n_in,
                              void* d_out, int out_size, void* d_ws, size_t ws_size,
                              hipStream_t stream) {
    (void)in_sizes; (void)n_in; (void)out_size; (void)ws_size;
    const float* vecs = (const float*)d_in[0];   // [16,32,128,64] fp32
    const float* cb   = (const float*)d_in[1];   // [512,64] fp32
    float* out = (float*)d_out;
    float* wsf = (float*)d_ws;
    float* cnorm2  = wsf;          // 512 floats
    float* partial = wsf + SCB;    // NWAVES floats

    const size_t ldsBytes = (size_t)SCB * KDIM * 2 * sizeof(__bf16) + SCB * sizeof(float);
    (void)hipFuncSetAttribute((const void*)vq_main,
                              hipFuncAttributeMaxDynamicSharedMemorySize, (int)ldsBytes);

    vq_prep<<<2, 256, 0, stream>>>(cb, cnorm2);
    vq_main<<<NQ / QPB, BLOCK, ldsBytes, stream>>>(vecs, cb, cnorm2, partial,
                                                   out + OFF_VHAT, out + OFF_Z);
    vq_fin<<<1, 32, 0, stream>>>(partial, out);
}
